// QConvIntNet_70557722738776
// MI455X (gfx1250) — matrix operand order repaired and checked by reading, not compiled, not run on hardware
//
#include <hip/hip_runtime.h>
#include <hip/hip_bf16.h>

// ---------------------------------------------------------------------------
// QConvIntNet on MI455X (gfx1250): all matmuls via v_wmma_f32_16x16x32_f16.
// Quantized weights (k/128) and qrelu activations (k/256) are exact in f16,
// so f16 WMMA with f32 accumulation reproduces the quantized math.
// ---------------------------------------------------------------------------

typedef __attribute__((ext_vector_type(16))) _Float16 v16h;
typedef __attribute__((ext_vector_type(8)))  float    v8f;

#define NCONST 128
#define NFEATS 16
#define BATCH  128

__device__ __forceinline__ float qbits_f(float x) {
    // quantized_bits(8,0,alpha=1): clamp to [-1, 127/128], round to 1/128 grid
    float y = fminf(fmaxf(x, -1.0f), 1.0f - 1.0f / 128.0f);
    return rintf(y * 128.0f) * (1.0f / 128.0f);
}

__device__ __forceinline__ float qrelu_f(float x) {
    // quantized_relu(8,0): clamp to [0, 255/256], round to 1/256 grid
    float y = fminf(fmaxf(x, 0.0f), 1.0f - 1.0f / 256.0f);
    return rintf(y * 256.0f) * (1.0f / 256.0f);
}

__device__ __forceinline__ v8f wmma_f16(v16h a, v16h b, v8f c) {
    // D = A(16x32 f16) * B(32x16 f16) + C(16x16 f32)
    return __builtin_amdgcn_wmma_f32_16x16x32_f16(
        /*neg_a=*/false, a, /*neg_b=*/false, b,
        /*c_mod=*/(short)0, c, /*reuse_a=*/false, /*reuse_b=*/false);
}

// Build a B operand (32x16 tile of a KxN row-major f32 weight matrix, qbits
// applied, zero-padded out of range). B layout (16-bit, wave32):
//   lanes 0-15  (h=0): N = lane;    VGPR 0-3: K=0..7,  VGPR 4-7: K=16..23
//   lanes 16-31 (h=1): N = lane-16; VGPR 0-3: K=8..15, VGPR 4-7: K=24..31
__device__ __forceinline__ v16h make_Bq(const float* W, int K, int N,
                                        int kbase, int nbase) {
    int lane = threadIdx.x & 31;
    int n = (lane & 15) + nbase;
    int h = lane >> 4;
    v16h b;
#pragma unroll
    for (int v = 0; v < 8; ++v) {
        int k = kbase + ((v < 4) ? (2 * v + 8 * h) : (16 + 2 * (v - 4) + 8 * h));
        float w0 = (k     < K && n < N) ? qbits_f(W[(size_t)k * N + n])       : 0.0f;
        float w1 = (k + 1 < K && n < N) ? qbits_f(W[(size_t)(k + 1) * N + n]) : 0.0f;
        b[2 * v]     = (_Float16)w0;
        b[2 * v + 1] = (_Float16)w1;
    }
    return b;
}

// Build an A operand (16x32) from a row-major f16 LDS buffer with leading
// dim ldh, rows starting at mbase, K columns starting at kbase.
// A layout (16-bit, wave32):
//   lanes 0-15  (h=0): M = lane;    VGPR 0-3: K=0..7,  VGPR 4-7: K=16..23
//   lanes 16-31 (h=1): M = lane-16; VGPR 0-3: K=8..15, VGPR 4-7: K=24..31
__device__ __forceinline__ v16h make_Ah(const _Float16* H, int ldh,
                                        int mbase, int kbase) {
    int lane = threadIdx.x & 31;
    int m = mbase + (lane & 15);
    int h = lane >> 4;
    const _Float16* row = H + (size_t)m * ldh;
    v16h a;
#pragma unroll
    for (int v = 0; v < 8; ++v) {
        int k = kbase + ((v < 4) ? (2 * v + 8 * h) : (16 + 2 * (v - 4) + 8 * h));
        a[2 * v]     = row[k];
        a[2 * v + 1] = row[k + 1];
    }
    return a;
}

// ---------------------------------------------------------------------------
// Kernel 1: edge MLP + receiver aggregation.
// Grid: 2048 blocks = 128 batches x 16 receiver-groups; 8 waves/block,
// wave w owns receiver r = 8*rg + w. Edges padded to 128 senders/receiver
// (diagonal s==r masked out of the aggregation) -> 8 tiles of 16 edges.
// ---------------------------------------------------------------------------
__global__ void __launch_bounds__(256)
edge_kernel(const float* __restrict__ x,
            const float* __restrict__ bn_gamma, const float* __restrict__ bn_beta,
            const float* __restrict__ bn_mean,  const float* __restrict__ bn_var,
            const float* __restrict__ fr_w1, const float* __restrict__ fr_b1,
            const float* __restrict__ fr_w2, const float* __restrict__ fr_b2,
            const float* __restrict__ fr_w3, const float* __restrict__ fr_b3,
            float* __restrict__ eff /* [B,128,6] */) {
    __shared__ _Float16 bnL[NCONST * NFEATS];   // 4 KB: bn for this batch (f16)
    __shared__ _Float16 h1s[8][16 * 32];        // 8 KB: layer-1 activations
    __shared__ _Float16 h2s[8][16 * 32];        // 8 KB: layer-2 activations

    const int b    = blockIdx.x >> 4;
    const int rg   = blockIdx.x & 15;
    const int wave = threadIdx.x >> 5;
    const int lane = threadIdx.x & 31;
    const int r    = rg * 8 + wave;

    // Cooperative BN -> LDS (f16)
    for (int i = threadIdx.x; i < NCONST * NFEATS; i += blockDim.x) {
        int f = i & (NFEATS - 1);
        float inv = rsqrtf(bn_var[f] + 1e-3f);
        float v = (x[(size_t)b * NCONST * NFEATS + i] - bn_mean[f]) * inv * bn_gamma[f] + bn_beta[f];
        bnL[i] = (_Float16)v;
    }
    __syncthreads();

    // Quantized weight operands (built once per wave)
    const v16h B1lo = make_Bq(fr_w1, 32, 30, 0,  0);   // fr_w1 cols 0-15
    const v16h B1hi = make_Bq(fr_w1, 32, 30, 0, 16);   // fr_w1 cols 16-29 (pad)
    const v16h B2   = make_Bq(fr_w2, 30, 15, 0,  0);   // fr_w2 (pad K->32, N->16)
    const v16h B3   = make_Bq(fr_w3, 15,  6, 0,  0);   // fr_w3 (pad K->32, N->16)

    const int n = lane & 15;          // column of C/D held by this lane
    const int h = lane >> 4;          // row-half of C/D (M = v + 8*h)
    const float b1a = qbits_f(fr_b1[n]);
    const float b1b = (16 + n < 30) ? qbits_f(fr_b1[16 + n]) : 0.0f;
    const float b2  = (n < 15) ? qbits_f(fr_b2[n]) : 0.0f;
    const float b3  = (n < 6)  ? qbits_f(fr_b3[n]) : 0.0f;

    _Float16* H1 = h1s[wave];
    _Float16* H2 = h2s[wave];
    // Pre-zero H2 columns 16..31 (K padding for layer 3); loop never writes them.
    for (int i = lane; i < 16 * 32; i += 32) H2[i] = (_Float16)0.0f;

    float acc = 0.0f;   // lane n (and n+16) accumulate effect column n over its M-half

    const _Float16* bnR = bnL + r * NFEATS;
    for (int t = 0; t < 8; ++t) {
        const int tile_base = t * 16;
        // --- A1: 16 edges x 32 feats = [bn[r] ++ bn[s_m]] ---
        v16h a1;
        {
            const int m = lane & 15;
            const _Float16* bnS = bnL + (tile_base + m) * NFEATS;
#pragma unroll
            for (int v = 0; v < 4; ++v) {
                int k = 2 * v + 8 * h;
                a1[2 * v]     = bnR[k];
                a1[2 * v + 1] = bnR[k + 1];
            }
#pragma unroll
            for (int v = 4; v < 8; ++v) {
                int k = 2 * (v - 4) + 8 * h;
                a1[2 * v]     = bnS[k];
                a1[2 * v + 1] = bnS[k + 1];
            }
        }
        v8f zero = {};
        // --- layer 1: 32 -> 30 (two N=16 tiles) ---
        v8f c0 = wmma_f16(a1, B1lo, zero);
        v8f c1 = wmma_f16(a1, B1hi, zero);
#pragma unroll
        for (int v = 0; v < 8; ++v) {
            int m = v + 8 * h;
            H1[m * 32 + n]      = (_Float16)qrelu_f(c0[v] + b1a);
            H1[m * 32 + 16 + n] = (_Float16)qrelu_f(c1[v] + b1b);  // cols 30,31 -> 0
        }
        // --- layer 2: 30(pad32) -> 15 ---
        v16h a2 = make_Ah(H1, 32, 0, 0);
        v8f c2 = wmma_f16(a2, B2, zero);
#pragma unroll
        for (int v = 0; v < 8; ++v) {
            int m = v + 8 * h;
            H2[m * 32 + n] = (_Float16)qrelu_f(c2[v] + b2);  // col 15 -> 0 (padded)
        }
        // --- layer 3: 15(pad32) -> 6, aggregate over senders (skip s==r) ---
        v16h a3 = make_Ah(H2, 32, 0, 0);
        v8f c3 = wmma_f16(a3, B3, zero);
        if (n < 6) {
#pragma unroll
            for (int v = 0; v < 8; ++v) {
                int m = v + 8 * h;
                float e = qrelu_f(c3[v] + b3);
                acc += (tile_base + m == r) ? 0.0f : e;
            }
        }
    }

    // combine the two row-halves (lanes n and n+16)
    acc += __shfl_xor(acc, 16, 32);
    if (lane < 6) eff[((size_t)b * NCONST + r) * 6 + lane] = acc;
}

// ---------------------------------------------------------------------------
// Kernel 2: node MLP fo (22->45->22->6) + per-tile node-sum partials.
// Grid: 128 blocks (one batch each), 8 waves, wave w owns nodes 16w..16w+15.
// ---------------------------------------------------------------------------
__global__ void __launch_bounds__(256)
node_kernel(const float* __restrict__ x,
            const float* __restrict__ bn_gamma, const float* __restrict__ bn_beta,
            const float* __restrict__ bn_mean,  const float* __restrict__ bn_var,
            const float* __restrict__ eff /* [B,128,6] */,
            const float* __restrict__ fo_w1, const float* __restrict__ fo_b1,
            const float* __restrict__ fo_w2, const float* __restrict__ fo_b2,
            const float* __restrict__ fo_w3, const float* __restrict__ fo_b3,
            float* __restrict__ partial /* [B,8,6] */) {
    __shared__ _Float16 AinL[NCONST * 32];   // 8 KB: [node][bn(16) ++ eff(6) ++ 0...]
    __shared__ _Float16 h1s[8][16 * 64];     // 16 KB: layer-1 activations (45 pad 64)
    __shared__ _Float16 h2s[8][16 * 32];     // 8 KB: layer-2 activations (22 pad 32)

    const int b    = blockIdx.x;
    const int wave = threadIdx.x >> 5;
    const int lane = threadIdx.x & 31;

    for (int i = threadIdx.x; i < NCONST * 32; i += blockDim.x) {
        int node = i >> 5;
        int c = i & 31;
        float v = 0.0f;
        if (c < 16) {
            float inv = rsqrtf(bn_var[c] + 1e-3f);
            v = (x[((size_t)b * NCONST + node) * NFEATS + c] - bn_mean[c]) * inv * bn_gamma[c] + bn_beta[c];
        } else if (c < 22) {
            v = eff[((size_t)b * NCONST + node) * 6 + (c - 16)];
        }
        AinL[i] = (_Float16)v;
    }
    __syncthreads();

    // Weight operands
    const v16h B1t0 = make_Bq(fo_w1, 22, 45, 0,  0);
    const v16h B1t1 = make_Bq(fo_w1, 22, 45, 0, 16);
    const v16h B1t2 = make_Bq(fo_w1, 22, 45, 0, 32);
    const v16h B2t00 = make_Bq(fo_w2, 45, 22,  0,  0);   // K rows 0-31,  cols 0-15
    const v16h B2t10 = make_Bq(fo_w2, 45, 22, 32,  0);   // K rows 32-44, cols 0-15
    const v16h B2t01 = make_Bq(fo_w2, 45, 22,  0, 16);   // K rows 0-31,  cols 16-21
    const v16h B2t11 = make_Bq(fo_w2, 45, 22, 32, 16);   // K rows 32-44, cols 16-21
    const v16h B3    = make_Bq(fo_w3, 22,  6,  0,  0);

    const int n = lane & 15;
    const int h = lane >> 4;
    const float b1t0 = (n      < 45) ? qbits_f(fo_b1[n])      : 0.0f;
    const float b1t1 = (16 + n < 45) ? qbits_f(fo_b1[16 + n]) : 0.0f;
    const float b1t2 = (32 + n < 45) ? qbits_f(fo_b1[32 + n]) : 0.0f;
    const float b2lo = (n      < 22) ? qbits_f(fo_b2[n])      : 0.0f;
    const float b2hi = (16 + n < 22) ? qbits_f(fo_b2[16 + n]) : 0.0f;
    const float b3n  = (n < 6)       ? qbits_f(fo_b3[n])      : 0.0f;

    _Float16* H1 = h1s[wave];
    _Float16* H2 = h2s[wave];
    for (int i = lane; i < 16 * 64; i += 32) H1[i] = (_Float16)0.0f;  // zero K-pad 45..63

    v8f zero = {};
    // --- layer 1: 22(pad32) -> 45 (3 N-tiles) ---
    v16h a1 = make_Ah(AinL, 32, wave * 16, 0);
    v8f d0 = wmma_f16(a1, B1t0, zero);
    v8f d1 = wmma_f16(a1, B1t1, zero);
    v8f d2 = wmma_f16(a1, B1t2, zero);
#pragma unroll
    for (int v = 0; v < 8; ++v) {
        int m = v + 8 * h;
        H1[m * 64 + n]      = (_Float16)qrelu_f(d0[v] + b1t0);
        H1[m * 64 + 16 + n] = (_Float16)qrelu_f(d1[v] + b1t1);
        H1[m * 64 + 32 + n] = (_Float16)qrelu_f(d2[v] + b1t2);  // cols 45-47 -> 0
    }
    // --- layer 2: 45(pad64) -> 22 (K chained through C operand, 2 N-tiles) ---
    v16h a20 = make_Ah(H1, 64, 0, 0);
    v16h a21 = make_Ah(H1, 64, 0, 32);
    v8f e0 = wmma_f16(a20, B2t00, zero);
    e0     = wmma_f16(a21, B2t10, e0);
    v8f e1 = wmma_f16(a20, B2t01, zero);
    e1     = wmma_f16(a21, B2t11, e1);
#pragma unroll
    for (int v = 0; v < 8; ++v) {
        int m = v + 8 * h;
        H2[m * 32 + n]      = (_Float16)qrelu_f(e0[v] + b2lo);
        H2[m * 32 + 16 + n] = (_Float16)qrelu_f(e1[v] + b2hi);  // cols 22-31 -> 0
    }
    // --- layer 3: 22(pad32) -> 6, node-sum over the 16 rows ---
    v16h a3 = make_Ah(H2, 32, 0, 0);
    v8f g = wmma_f16(a3, B3, zero);
    float acc = 0.0f;
    if (n < 6) {
#pragma unroll
        for (int v = 0; v < 8; ++v) acc += qrelu_f(g[v] + b3n);
    }
    acc += __shfl_xor(acc, 16, 32);
    if (lane < 6) partial[((size_t)b * 8 + wave) * 6 + lane] = acc;
}

// ---------------------------------------------------------------------------
// Kernel 3: pool partials, fc head, softmax. One thread per batch.
// ---------------------------------------------------------------------------
__global__ void head_kernel(const float* __restrict__ partial /* [B,8,6] */,
                            const float* __restrict__ fc_w1 /* [6,48] */,
                            const float* __restrict__ fc_b1 /* [48]  */,
                            const float* __restrict__ fc_w2 /* [48,5] */,
                            const float* __restrict__ fc_b2 /* [5]   */,
                            float* __restrict__ out /* [B,5] */) {
    int b = blockIdx.x * blockDim.x + threadIdx.x;
    if (b >= BATCH) return;

    float pooled[6];
#pragma unroll
    for (int k = 0; k < 6; ++k) {
        float s = 0.0f;
        for (int w = 0; w < 8; ++w) s += partial[((size_t)b * 8 + w) * 6 + k];
        pooled[k] = s;
    }
    float c[48];
    for (int j = 0; j < 48; ++j) {
        float s = qbits_f(fc_b1[j]);
#pragma unroll
        for (int k = 0; k < 6; ++k) s += pooled[k] * qbits_f(fc_w1[k * 48 + j]);
        c[j] = qrelu_f(s);
    }
    float logit[5], mx = -1e30f;
#pragma unroll
    for (int j = 0; j < 5; ++j) {
        float s = qbits_f(fc_b2[j]);
        for (int k = 0; k < 48; ++k) s += c[k] * qbits_f(fc_w2[k * 5 + j]);
        logit[j] = s;
        mx = fmaxf(mx, s);
    }
    float denom = 0.0f;
#pragma unroll
    for (int j = 0; j < 5; ++j) { logit[j] = __expf(logit[j] - mx); denom += logit[j]; }
    float inv = 1.0f / denom;
#pragma unroll
    for (int j = 0; j < 5; ++j) out[(size_t)b * 5 + j] = logit[j] * inv;
}

extern "C" void kernel_launch(void* const* d_in, const int* in_sizes, int n_in,
                              void* d_out, int out_size, void* d_ws, size_t ws_size,
                              hipStream_t stream) {
    (void)in_sizes; (void)n_in; (void)out_size; (void)ws_size;
    const float* x        = (const float*)d_in[0];
    const float* bn_gamma = (const float*)d_in[1];
    const float* bn_beta  = (const float*)d_in[2];
    const float* bn_mean  = (const float*)d_in[3];
    const float* bn_var   = (const float*)d_in[4];
    const float* fr_w1 = (const float*)d_in[5];
    const float* fr_b1 = (const float*)d_in[6];
    const float* fr_w2 = (const float*)d_in[7];
    const float* fr_b2 = (const float*)d_in[8];
    const float* fr_w3 = (const float*)d_in[9];
    const float* fr_b3 = (const float*)d_in[10];
    const float* fo_w1 = (const float*)d_in[11];
    const float* fo_b1 = (const float*)d_in[12];
    const float* fo_w2 = (const float*)d_in[13];
    const float* fo_b2 = (const float*)d_in[14];
    const float* fo_w3 = (const float*)d_in[15];
    const float* fo_b3 = (const float*)d_in[16];
    const float* fc_w1 = (const float*)d_in[17];
    const float* fc_b1 = (const float*)d_in[18];
    const float* fc_w2 = (const float*)d_in[19];
    const float* fc_b2 = (const float*)d_in[20];

    float* eff     = (float*)d_ws;                     // [B,128,6]  = 98304 floats
    float* partial = eff + (size_t)BATCH * NCONST * 6; // [B,8,6]    = 6144 floats

    edge_kernel<<<BATCH * 16, 256, 0, stream>>>(
        x, bn_gamma, bn_beta, bn_mean, bn_var,
        fr_w1, fr_b1, fr_w2, fr_b2, fr_w3, fr_b3, eff);

    node_kernel<<<BATCH, 256, 0, stream>>>(
        x, bn_gamma, bn_beta, bn_mean, bn_var, eff,
        fo_w1, fo_b1, fo_w2, fo_b2, fo_w3, fo_b3, partial);

    head_kernel<<<1, BATCH, 0, stream>>>(
        partial, fc_w1, fc_b1, fc_w2, fc_b2, (float*)d_out);
}
